// GeniePath_6408091205711
// MI455X (gfx1250) — compile-verified, matrix operand order checked by reading, not run on hardware
//
#include <hip/hip_runtime.h>
#include <hip/hip_bf16.h>

// ---------------------------------------------------------------------------
// GeniePath on MI455X (gfx1250): f16 WMMA GEMMs, fully fused WMMA LSTM cell,
// atomic edge softmax/scatter.
// ---------------------------------------------------------------------------

#define NN   30000      // nodes
#define EE   120000     // edges (before self loops)
#define FF   14         // input features
#define KPAD 32         // F padded to WMMA K step
#define DD   256        // hidden dim
#define G4   1024       // 4*D LSTM gates
#define ETOT (EE + NN)  // edges + self loops
#define NEG_SLOPE 0.2f

typedef _Float16 v8h  __attribute__((ext_vector_type(8)));
typedef _Float16 v16h __attribute__((ext_vector_type(16)));
typedef float    v8f  __attribute__((ext_vector_type(8)));

// Load a 16-half WMMA fragment: two contiguous 16B chunks at +0 and +16 halves
// (ISA 16-bit A/B layout: lane holds 8 K-values at khalf*8 in v0..3 and 8 more
// at +16 in v4..7).
#define LOAD_FRAG(dst, ptr)                                    \
  do {                                                         \
    union { v16h v; v8h h[2]; } _u;                            \
    _u.h[0] = *(const v8h*)((ptr));                            \
    _u.h[1] = *(const v8h*)((ptr) + 16);                       \
    (dst) = _u.v;                                              \
  } while (0)

#define WMMA(a, b, c)                                                        \
  __builtin_amdgcn_wmma_f32_16x16x32_f16(false, (a), false, (b), (short)0,   \
                                         (c), false, false)

// ---------------------------------------------------------------------------
// GEMM: out[M,N] = A[M,K](f16,row) @ W[N,K](f16,row)^T (+bias).
// Writes f32 (C) and/or f16 (C16) per non-null pointer. K % 32 == 0.
// Block = 256 threads = 8 waves; tile 128(M) x 64(N); wave: 16x64 (A reuse x4)
// ---------------------------------------------------------------------------
__global__ __launch_bounds__(256) void gemm_f16_wmma(
    const _Float16* __restrict__ A, const _Float16* __restrict__ B,
    float* __restrict__ C, _Float16* __restrict__ C16,
    const float* __restrict__ bias, int M, int N, int K) {
  const int tid    = threadIdx.x;
  const int w      = tid >> 5;
  const int l      = tid & 31;
  const int lane16 = l & 15;
  const int lhalf  = l >> 4;

  const int m0 = blockIdx.x * 128 + w * 16;
  const int n0 = blockIdx.y * 64;

  const int  am      = m0 + lane16;
  const bool arow_ok = (am < M);
  const _Float16* Arow  = A + (size_t)am * K;
  const _Float16* Bbase = B + (size_t)(n0 + lane16) * K;

  v8f acc0 = {}, acc1 = {}, acc2 = {}, acc3 = {};

  for (int kk = 0; kk < K; kk += 32) {
    const int kb = kk + lhalf * 8;
    v16h a = {};
    if (arow_ok) LOAD_FRAG(a, Arow + kb);
    v16h b0, b1, b2, b3;
    LOAD_FRAG(b0, Bbase + (size_t)0  * K + kb);
    LOAD_FRAG(b1, Bbase + (size_t)16 * K + kb);
    LOAD_FRAG(b2, Bbase + (size_t)32 * K + kb);
    LOAD_FRAG(b3, Bbase + (size_t)48 * K + kb);
    acc0 = WMMA(a, b0, acc0);
    acc1 = WMMA(a, b1, acc1);
    acc2 = WMMA(a, b2, acc2);
    acc3 = WMMA(a, b3, acc3);
  }

  // C/D layout: lane l -> N = n0 + t*16 + (l&15); VGPR j -> M = (l>>4)*8 + j.
#pragma unroll
  for (int j = 0; j < 8; ++j) {
    const int m = m0 + lhalf * 8 + j;
    if (m >= M) continue;
    const size_t rowb = (size_t)m * N;
#define EPILOG(t, accv)                                        \
    {                                                          \
      const int n  = n0 + (t) * 16 + lane16;                   \
      float v = (accv)[j];                                     \
      if (bias) v += bias[n];                                  \
      if (C)   C[rowb + n]   = v;                              \
      if (C16) C16[rowb + n] = (_Float16)v;                    \
    }
    EPILOG(0, acc0)
    EPILOG(1, acc1)
    EPILOG(2, acc2)
    EPILOG(3, acc3)
#undef EPILOG
  }
}

// ---------------------------------------------------------------------------
// Fully fused LSTM cell:
//   gates = Ab @ Wih^T + Ah @ Whh^T   (never materialized)
//   c' = sig(f)*c + sig(i)*tanh(g);  h' = sig(o)*tanh(c')
// Wave owns one 16x16 tile of D, with its 4 accumulators covering the 4 gate
// column-blocks (n, n+D, n+2D, n+3D) -> i/f/g/o live in registers at epilogue.
// Block = 8 waves = 32(M) x 64(N of D). Hout must not alias Ab/Ah.
// ---------------------------------------------------------------------------
__global__ __launch_bounds__(256) void lstm_fused_wmma(
    const _Float16* __restrict__ Ab,   // [NN, DD] GAT output (tanh'd)
    const _Float16* __restrict__ Ah,   // [NN, DD] previous hidden
    const _Float16* __restrict__ Wih,  // [G4, DD]
    const _Float16* __restrict__ Whh,  // [G4, DD]
    float* __restrict__ cs,            // [NN, DD] cell state (in-place)
    _Float16* __restrict__ Hout) {     // [NN, DD] new hidden
  const int tid    = threadIdx.x;
  const int w      = tid >> 5;
  const int wm     = w >> 2;   // 0..1
  const int wn     = w & 3;    // 0..3
  const int l      = tid & 31;
  const int lane16 = l & 15;
  const int lhalf  = l >> 4;

  const int m0 = blockIdx.x * 32 + wm * 16;
  const int n0 = blockIdx.y * 64 + wn * 16;   // column within D

  const int  am      = m0 + lane16;
  const bool arow_ok = (am < NN);
  const _Float16* Abrow = Ab + (size_t)am * DD;
  const _Float16* Ahrow = Ah + (size_t)am * DD;
  const int brow = n0 + lane16;               // B row within a gate block

  v8f acc_i = {}, acc_f = {}, acc_g = {}, acc_o = {};

  for (int kk = 0; kk < DD; kk += 32) {
    const int kb = kk + lhalf * 8;
    v16h ab = {}, ah = {};
    if (arow_ok) {
      LOAD_FRAG(ab, Abrow + kb);
      LOAD_FRAG(ah, Ahrow + kb);
    }
#define GATE(g, accv)                                                       \
    {                                                                       \
      v16h bw, bh;                                                          \
      LOAD_FRAG(bw, Wih + (size_t)((g) * DD + brow) * DD + kb);             \
      LOAD_FRAG(bh, Whh + (size_t)((g) * DD + brow) * DD + kb);             \
      accv = WMMA(ab, bw, accv);                                            \
      accv = WMMA(ah, bh, accv);                                            \
    }
    GATE(0, acc_i)
    GATE(1, acc_f)
    GATE(2, acc_g)
    GATE(3, acc_o)
#undef GATE
  }

#pragma unroll
  for (int j = 0; j < 8; ++j) {
    const int m = m0 + lhalf * 8 + j;
    if (m >= NN) continue;
    const size_t idx = (size_t)m * DD + n0 + lane16;
    float ig = 1.0f / (1.0f + __expf(-acc_i[j]));
    float fg = 1.0f / (1.0f + __expf(-acc_f[j]));
    float gg = tanhf(acc_g[j]);
    float og = 1.0f / (1.0f + __expf(-acc_o[j]));
    float c  = fg * cs[idx] + ig * gg;
    cs[idx]  = c;
    Hout[idx] = (_Float16)(og * tanhf(c));
  }
}

// ---------------------------------------------------------------------------
// Utility kernels
// ---------------------------------------------------------------------------
__global__ void zero_u32_kernel(unsigned* __restrict__ p, long long n) {
  long long i = (long long)blockIdx.x * blockDim.x + threadIdx.x;
  if (i < n) p[i] = 0u;
}

__global__ void cvt_f32_f16_kernel(const float* __restrict__ s,
                                   _Float16* __restrict__ d, long long n) {
  long long i = (long long)blockIdx.x * blockDim.x + threadIdx.x;
  if (i < n) d[i] = (_Float16)s[i];
}

// rows x FF (f32) -> rows x KPAD (f16, zero padded)
__global__ void pad_cvt_kernel(const float* __restrict__ s,
                               _Float16* __restrict__ d, int rows) {
  long long i = (long long)blockIdx.x * blockDim.x + threadIdx.x;
  if (i < (long long)rows * KPAD) {
    int r = (int)(i >> 5), c = (int)(i & (KPAD - 1));
    d[i] = (c < FF) ? (_Float16)s[(size_t)r * FF + c] : (_Float16)0.0f;
  }
}

// per-node attention scores: e = hw . a_src, d = hw . a_dst  (1 wave / node)
__global__ void scores_kernel(const float* __restrict__ hw,
                              const float* __restrict__ asrc,
                              const float* __restrict__ adst,
                              float* __restrict__ e, float* __restrict__ d) {
  int node = blockIdx.x * 8 + (threadIdx.x >> 5);
  int lane = threadIdx.x & 31;
  if (node >= NN) return;
  const float* row = hw + (size_t)node * DD;
  float se = 0.f, sd = 0.f;
#pragma unroll
  for (int i = 0; i < 8; ++i) {
    int k = lane + 32 * i;
    float v = row[k];
    se += v * asrc[k];
    sd += v * adst[k];
  }
#pragma unroll
  for (int o = 16; o > 0; o >>= 1) {
    se += __shfl_xor(se, o, 32);
    sd += __shfl_xor(sd, o, 32);
  }
  if (lane == 0) { e[node] = se; d[node] = sd; }
}

__device__ __forceinline__ unsigned f32_order_key(float f) {
  unsigned u = __float_as_uint(f);
  return (u & 0x80000000u) ? ~u : (u | 0x80000000u);
}
__device__ __forceinline__ float f32_order_unkey(unsigned k) {
  return (k & 0x80000000u) ? __uint_as_float(k & 0x7FFFFFFFu)
                           : __uint_as_float(~k);
}

__device__ __forceinline__ void edge_endpoints(const long long* ei, int idx,
                                               int& s, int& t) {
  if (idx < EE) { s = (int)ei[idx]; t = (int)ei[EE + idx]; }
  else          { s = t = idx - EE; }   // self loop
}

// leaky-relu logit + segment max (order-preserving uint atomicMax)
__global__ void edge_logit_max_kernel(const long long* __restrict__ ei,
                                      const float* __restrict__ e,
                                      const float* __restrict__ d,
                                      float* __restrict__ logit,
                                      unsigned* __restrict__ mkey) {
  int idx = blockIdx.x * blockDim.x + threadIdx.x;
  if (idx >= ETOT) return;
  int s, t; edge_endpoints(ei, idx, s, t);
  float lg = e[s] + d[t];
  lg = (lg > 0.f) ? lg : NEG_SLOPE * lg;
  logit[idx] = lg;
  atomicMax(&mkey[t], f32_order_key(lg));
}

// exp(logit - max) + segment sum
__global__ void edge_exp_sum_kernel(const long long* __restrict__ ei,
                                    const float* __restrict__ logit,
                                    const unsigned* __restrict__ mkey,
                                    float* __restrict__ ex,
                                    float* __restrict__ z) {
  int idx = blockIdx.x * blockDim.x + threadIdx.x;
  if (idx >= ETOT) return;
  int s, t; edge_endpoints(ei, idx, s, t);
  (void)s;
  float v = __expf(logit[idx] - f32_order_unkey(mkey[t]));
  ex[idx] = v;
  atomicAdd(&z[t], v);
}

// agg[dst] += (ex/z[dst]) * hw[src]   (1 wave / edge, lane-strided over D)
__global__ void edge_agg_kernel(const long long* __restrict__ ei,
                                const float* __restrict__ ex,
                                const float* __restrict__ z,
                                const float* __restrict__ hw,
                                float* __restrict__ agg) {
  int idx = (blockIdx.x * blockDim.x + threadIdx.x) >> 5;
  if (idx >= ETOT) return;
  int s, t; edge_endpoints(ei, idx, s, t);
  float alpha = ex[idx] / z[t];
  int lane = threadIdx.x & 31;
  const float* srow = hw + (size_t)s * DD;
  float* drow = agg + (size_t)t * DD;
#pragma unroll
  for (int i = 0; i < 8; ++i) {
    int k = lane + 32 * i;
    atomicAdd(&drow[k], alpha * srow[k]);
  }
}

// b16 = f16(tanh(agg + gat_b))
__global__ void tanh_bias_cvt_kernel(const float* __restrict__ agg,
                                     const float* __restrict__ gb,
                                     _Float16* __restrict__ b16) {
  long long i = (long long)blockIdx.x * blockDim.x + threadIdx.x;
  if (i < (long long)NN * DD) {
    int dim = (int)(i & (DD - 1));
    b16[i] = (_Float16)tanhf(agg[i] + gb[dim]);
  }
}

// ---------------------------------------------------------------------------
// Host-side orchestration
// ---------------------------------------------------------------------------
static inline dim3 blocks1d(long long n, int bs) {
  return dim3((unsigned)((n + bs - 1) / bs));
}

extern "C" void kernel_launch(void* const* d_in, const int* in_sizes, int n_in,
                              void* d_out, int out_size, void* d_ws,
                              size_t ws_size, hipStream_t stream) {
  (void)in_sizes; (void)n_in; (void)out_size; (void)ws_size;

  const float*     x      = (const float*)d_in[0];
  const long long* ei     = (const long long*)d_in[1];
  const float*     lin1_w = (const float*)d_in[2];
  const float*     lin1_b = (const float*)d_in[3];
  const float*     lin2_w = (const float*)d_in[4];
  const float*     lin2_b = (const float*)d_in[5];
  const float *gat_w[4], *att_s[4], *att_d[4], *gat_b[4], *wih[4], *whh[4];
  for (int i = 0; i < 4; ++i) {
    int b = 6 + i * 6;
    gat_w[i] = (const float*)d_in[b + 0];
    att_s[i] = (const float*)d_in[b + 1];
    att_d[i] = (const float*)d_in[b + 2];
    gat_b[i] = (const float*)d_in[b + 3];
    wih[i]   = (const float*)d_in[b + 4];
    whh[i]   = (const float*)d_in[b + 5];
  }
  float* out = (float*)d_out;

  // workspace bump allocator
  char* wp = (char*)d_ws;
  auto alloc = [&](size_t bytes) -> void* {
    void* p = (void*)wp;
    wp += (bytes + 255) & ~(size_t)255;
    return p;
  };
  _Float16* x16   = (_Float16*)alloc((size_t)NN * KPAD * 2);
  _Float16* w1_16 = (_Float16*)alloc((size_t)DD * KPAD * 2);
  _Float16* w2_16 = (_Float16*)alloc((size_t)DD * DD * 2);
  _Float16 *gw16[4], *wih16[4], *whh16[4];
  for (int i = 0; i < 4; ++i) {
    gw16[i]  = (_Float16*)alloc((size_t)DD * DD * 2);
    wih16[i] = (_Float16*)alloc((size_t)G4 * DD * 2);
    whh16[i] = (_Float16*)alloc((size_t)G4 * DD * 2);
  }
  _Float16* hA16 = (_Float16*)alloc((size_t)NN * DD * 2);
  _Float16* hB16 = (_Float16*)alloc((size_t)NN * DD * 2);
  _Float16* hz16 = (_Float16*)alloc((size_t)NN * DD * 2);  // zero hidden
  _Float16* b16  = (_Float16*)alloc((size_t)NN * DD * 2);
  float*    cs   = (float*)alloc((size_t)NN * DD * 4);
  float*    hw   = (float*)alloc((size_t)NN * DD * 4);
  float*    agg  = (float*)alloc((size_t)NN * DD * 4);
  float*    ev   = (float*)alloc((size_t)NN * 4);
  float*    dv   = (float*)alloc((size_t)NN * 4);
  unsigned* mkey = (unsigned*)alloc((size_t)NN * 4);
  float*    z    = (float*)alloc((size_t)NN * 4);
  float*    lgt  = (float*)alloc((size_t)ETOT * 4);
  float*    ex   = (float*)alloc((size_t)ETOT * 4);

  const int BS = 256;
  const long long ND = (long long)NN * DD;

  // ---- weight / input conversion to f16 ----
  pad_cvt_kernel<<<blocks1d((long long)NN * KPAD, BS), BS, 0, stream>>>(x, x16, NN);
  pad_cvt_kernel<<<blocks1d((long long)DD * KPAD, BS), BS, 0, stream>>>(lin1_w, w1_16, DD);
  cvt_f32_f16_kernel<<<blocks1d((long long)DD * DD, BS), BS, 0, stream>>>(lin2_w, w2_16, (long long)DD * DD);
  for (int i = 0; i < 4; ++i) {
    cvt_f32_f16_kernel<<<blocks1d((long long)DD * DD, BS), BS, 0, stream>>>(gat_w[i], gw16[i], (long long)DD * DD);
    cvt_f32_f16_kernel<<<blocks1d((long long)G4 * DD, BS), BS, 0, stream>>>(wih[i], wih16[i], (long long)G4 * DD);
    cvt_f32_f16_kernel<<<blocks1d((long long)G4 * DD, BS), BS, 0, stream>>>(whh[i], whh16[i], (long long)G4 * DD);
  }

  // ---- init state: h_state = 0 (f16), cs = 0 (f32) ----
  zero_u32_kernel<<<blocks1d(ND / 2, BS), BS, 0, stream>>>((unsigned*)hz16, ND / 2);
  zero_u32_kernel<<<blocks1d(ND, BS), BS, 0, stream>>>((unsigned*)cs, ND);

  const dim3 gemmB(256);
  const dim3 gD((NN + 127) / 128, DD / 64);      // generic GEMM, N = 256
  const dim3 gL((NN + 31) / 32, DD / 64);        // fused LSTM

  // ---- lin1: h = f16(x @ W1^T + b1)  (K padded to 32) ----
  gemm_f16_wmma<<<gD, gemmB, 0, stream>>>(x16, w1_16, nullptr, hA16, lin1_b,
                                          NN, DD, KPAD);

  // ---- 4 GeniePath layers (ping-pong hidden state) ----
  _Float16* hin  = hA16;
  _Float16* hout = hB16;
  for (int i = 0; i < 4; ++i) {
    // GAT linear projection: hw = h @ W^T  (f32, needed by softmax/aggregate)
    gemm_f16_wmma<<<gD, gemmB, 0, stream>>>(hin, gw16[i], hw, nullptr, nullptr,
                                            NN, DD, DD);
    // per-node scores
    scores_kernel<<<(NN + 7) / 8, 256, 0, stream>>>(hw, att_s[i], att_d[i], ev, dv);
    // reset segment buffers
    zero_u32_kernel<<<blocks1d(NN, BS), BS, 0, stream>>>(mkey, NN);
    zero_u32_kernel<<<blocks1d(NN, BS), BS, 0, stream>>>((unsigned*)z, NN);
    zero_u32_kernel<<<blocks1d(ND, BS), BS, 0, stream>>>((unsigned*)agg, ND);
    // edge softmax + aggregate
    edge_logit_max_kernel<<<blocks1d(ETOT, BS), BS, 0, stream>>>(ei, ev, dv, lgt, mkey);
    edge_exp_sum_kernel<<<blocks1d(ETOT, BS), BS, 0, stream>>>(ei, lgt, mkey, ex, z);
    edge_agg_kernel<<<blocks1d((long long)ETOT * 32, BS), BS, 0, stream>>>(ei, ex, z, hw, agg);
    // b = tanh(agg + gat_b) -> f16
    tanh_bias_cvt_kernel<<<blocks1d(ND, BS), BS, 0, stream>>>(agg, gat_b[i], b16);
    // fused LSTM cell: gates GEMMs + pointwise, no gates buffer
    const _Float16* hs = (i == 0) ? hz16 : hin;
    lstm_fused_wmma<<<gL, gemmB, 0, stream>>>(b16, hs, wih16[i], whh16[i], cs, hout);
    // h = hs = hout
    _Float16* t = hin; hin = hout; hout = t;
  }

  // ---- lin2: out = h @ W2^T + b2 ----
  gemm_f16_wmma<<<gD, gemmB, 0, stream>>>(hin, w2_16, out, nullptr, lin2_b,
                                          NN, DD, DD);
}
